// Evolution_84851373900514
// MI455X (gfx1250) — compile-verified, matrix operand order checked by reading, not run on hardware
//
#include <hip/hip_runtime.h>
#include <hip/hip_bf16.h>

typedef __attribute__((ext_vector_type(16))) __bf16 v16bf;
typedef __attribute__((ext_vector_type(8)))  __bf16 v8bf;
typedef __attribute__((ext_vector_type(8)))  float  v8f;

#define T_STEPS 64
#define NROWS   2048
#define DR2     512
#define KDIM    1024
#define BM      64
#define BN      64
#define BK      32
#define DECAY_F 0.6065306597126334f

// Prep: w1 [KDIM, DR2] f32 -> w1t [DR2, KDIM] bf16 (transposed so B fragments
// are contiguous-K per column), and copy h0 into all_data_dynamic[0].
__global__ __launch_bounds__(512) void prep_kernel(const float* __restrict__ w1,
                                                   const float* __restrict__ h0,
                                                   __bf16* __restrict__ w1t,
                                                   float* __restrict__ dyn0) {
  const size_t idx = (size_t)blockIdx.x * blockDim.x + threadIdx.x;
  if (idx < (size_t)DR2 * KDIM) {
    const int n = (int)(idx >> 10);          // output row (column of w1)
    const int k = (int)(idx & (KDIM - 1));   // K index
    w1t[(size_t)n * KDIM + k] = (__bf16)w1[(size_t)k * DR2 + n];
  }
  if (idx < (size_t)NROWS * DR2) {
    dyn0[idx] = h0[idx];
  }
}

// One recurrence step:  h_new = sigmoid( (concat(h,static)@w1)*thr + h*(1-thr) ) * DECAY
// GEMM tile: 64x64 per workgroup, 4 waves, each wave owns a 16-col strip and
// 4 row-tiles of 16x16 accumulated with v_wmma_f32_16x16x32_bf16.
// A tile is double-buffered in LDS: next chunk's f32 loads are issued before
// the current chunk's WMMAs so they complete under compute.
__global__ __launch_bounds__(128) void step_kernel(
    const float* __restrict__ hprev,    // [N, DR2] f32 (states[t-1] or h0)
    const float* __restrict__ stat_t,   // [N, DR2] f32 (static slice at t)
    const float* __restrict__ thr_t,    // [N] f32
    const __bf16* __restrict__ w1t,     // [DR2, KDIM] bf16 (transposed weights)
    float* __restrict__ hout,           // [N, DR2] states[t]
    float* __restrict__ diffout,        // [N, DR2] states[t]-states[t-1], or null
    float* __restrict__ finout) {       // [N, DR2] final copy, or null
  __shared__ __bf16 As[2][BM * BK];     // 2 x 4 KB staged A tiles (bf16)

  const int tid  = threadIdx.x;
  const int lane = tid & 31;
  const int wave = tid >> 5;
  const int m    = lane & 15;   // row (A) / col (B) within 16x16 tile
  const int hi   = lane >> 4;   // K-half select per documented 16-bit layout

  const int rowBase  = blockIdx.x * BM;
  const int colStrip = blockIdx.y * BN + wave * 16;

  v8f acc[4];
  #pragma unroll
  for (int i = 0; i < 4; ++i)
    acc[i] = (v8f){0.f, 0.f, 0.f, 0.f, 0.f, 0.f, 0.f, 0.f};

  // A-staging assignment: each of 128 threads converts 16 consecutive f32.
  const int sr = tid >> 1;        // tile row 0..63
  const int sk = (tid & 1) * 16;  // K half: 0 or 16
  const float* rowh = hprev  + (size_t)(rowBase + sr) * DR2;
  const float* rows = stat_t + (size_t)(rowBase + sr) * DR2;

  // ---- prologue: stage kc=0 (always from hprev half) into buffer 0 ----
  {
    const float4* s4 = reinterpret_cast<const float4*>(rowh + sk);
    __bf16 tmp[16];
    #pragma unroll
    for (int j = 0; j < 4; ++j) {
      const float4 f = s4[j];
      tmp[4 * j + 0] = (__bf16)f.x;
      tmp[4 * j + 1] = (__bf16)f.y;
      tmp[4 * j + 2] = (__bf16)f.z;
      tmp[4 * j + 3] = (__bf16)f.w;
    }
    v8bf lo, hi8;
    #pragma unroll
    for (int j = 0; j < 8; ++j) { lo[j] = tmp[j]; hi8[j] = tmp[8 + j]; }
    v8bf* dst = reinterpret_cast<v8bf*>(&As[0][sr * BK + sk]);
    dst[0] = lo;
    dst[1] = hi8;
  }
  __syncthreads();

  int p = 0;
  for (int kc = 0; kc < KDIM; kc += BK) {
    // ---- issue next A-chunk global loads early (uniform branch) ----
    const bool havenext = (kc + BK) < KDIM;
    float4 pre[4];
    if (havenext) {
      const int kg = kc + BK + sk;
      const float* src = (kg < DR2) ? (rowh + kg) : (rows + (kg - DR2));
      const float4* s4 = reinterpret_cast<const float4*>(src);
      #pragma unroll
      for (int j = 0; j < 4; ++j) pre[j] = s4[j];
    }

    // ---- B fragment straight from global (w1t is 1 MB -> L2 resident) ----
    // lane<16: col=m, K kc+0..7 & kc+16..23 ; lane>=16: K kc+8..15 & kc+24..31
    const __bf16* bp = w1t + (size_t)(colStrip + m) * KDIM + kc + hi * 8;
    const v8bf b0 = *reinterpret_cast<const v8bf*>(bp);
    const v8bf b1 = *reinterpret_cast<const v8bf*>(bp + 16);
    v16bf bfrag;
    #pragma unroll
    for (int j = 0; j < 8; ++j) { bfrag[j] = b0[j]; bfrag[8 + j] = b1[j]; }

    // ---- 4 row-tiles of WMMA from current buffer ----
    #pragma unroll
    for (int tr = 0; tr < 4; ++tr) {
      const __bf16* ap = &As[p][(tr * 16 + m) * BK + hi * 8];
      const v8bf a0 = *reinterpret_cast<const v8bf*>(ap);        // ds_load_b128
      const v8bf a1 = *reinterpret_cast<const v8bf*>(ap + 16);   // ds_load_b128
      v16bf afrag;
      #pragma unroll
      for (int j = 0; j < 8; ++j) { afrag[j] = a0[j]; afrag[8 + j] = a1[j]; }
      acc[tr] = __builtin_amdgcn_wmma_f32_16x16x32_bf16(
          false, afrag, false, bfrag, (short)0, acc[tr], false, false);
    }

    // ---- convert prefetched data and store into the other buffer ----
    if (havenext) {
      __bf16 tmp[16];
      #pragma unroll
      for (int j = 0; j < 4; ++j) {
        tmp[4 * j + 0] = (__bf16)pre[j].x;
        tmp[4 * j + 1] = (__bf16)pre[j].y;
        tmp[4 * j + 2] = (__bf16)pre[j].z;
        tmp[4 * j + 3] = (__bf16)pre[j].w;
      }
      v8bf lo, hi8;
      #pragma unroll
      for (int j = 0; j < 8; ++j) { lo[j] = tmp[j]; hi8[j] = tmp[8 + j]; }
      v8bf* dst = reinterpret_cast<v8bf*>(&As[p ^ 1][sr * BK + sk]);
      dst[0] = lo;
      dst[1] = hi8;
    }
    __syncthreads();
    p ^= 1;
  }

  // ---- epilogue: gated sigmoid recurrence, all outputs in one pass ----
  // C/D layout: VGPR e, lane l -> row = 16*tr + e + 8*(l>=16), col = l&15.
  // sigmoid via v_exp + v_rcp (hardware reciprocal; inputs already bf16-rounded).
  const int col = colStrip + m;
  #pragma unroll
  for (int tr = 0; tr < 4; ++tr) {
    #pragma unroll
    for (int e = 0; e < 8; ++e) {
      const int row   = rowBase + tr * 16 + e + 8 * hi;
      const float th  = thr_t[row];
      const size_t o  = (size_t)row * DR2 + col;
      const float hp  = hprev[o];
      const float z   = acc[tr][e] * th + hp * (1.0f - th);
      const float s   = DECAY_F * __builtin_amdgcn_rcpf(1.0f + __expf(-z));
      hout[o] = s;
      if (diffout) diffout[o] = s - hp;
      if (finout)  finout[o]  = s;
    }
  }
}

extern "C" void kernel_launch(void* const* d_in, const int* in_sizes, int n_in,
                              void* d_out, int out_size, void* d_ws, size_t ws_size,
                              hipStream_t stream) {
  const float* a_static = (const float*)d_in[0];  // [T, N, DR2]
  const float* thr      = (const float*)d_in[1];  // [T, N, 1]
  const float* h0       = (const float*)d_in[2];  // [N, DR2]
  const float* w1       = (const float*)d_in[3];  // [2*DR2, DR2]

  float* out = (float*)d_out;
  const size_t NDR = (size_t)NROWS * DR2;
  float* dyn  = out;                          // [T, N, DR2]
  float* fin  = out + (size_t)T_STEPS * NDR;  // [N, DR2]  (scratch for states[0] until t=63)
  float* diff = fin + NDR;                    // [T-1, N, DR2]

  __bf16* w1t = (__bf16*)d_ws;                // DR2*KDIM bf16 = 1 MB scratch

  {
    const int total = NROWS * DR2;  // covers both prep workloads
    prep_kernel<<<(total + 511) / 512, 512, 0, stream>>>(w1, h0, w1t, dyn);
  }

  const dim3 grid(NROWS / BM, DR2 / BN);  // 32 x 8 = 256 workgroups
  const dim3 block(128);
  for (int t = 0; t < T_STEPS; ++t) {
    const float* hprev = (t == 0) ? h0
                       : (t == 1) ? fin
                                  : dyn + (size_t)(t - 1) * NDR;
    float* hout = (t == 0) ? fin : dyn + (size_t)t * NDR;
    float* dout = (t >= 1) ? diff + (size_t)(t - 1) * NDR : nullptr;
    float* fout = (t == T_STEPS - 1) ? fin : nullptr;
    step_kernel<<<grid, block, 0, stream>>>(
        hprev, a_static + (size_t)t * NDR, thr + (size_t)t * NROWS,
        w1t, hout, dout, fout);
  }
}
